// MaskedEncoderLayer_44839458570555
// MI455X (gfx1250) — matrix operand order repaired and checked by reading, not compiled, not run on hardware
//
#include <hip/hip_runtime.h>

// ---------------- problem constants (match reference) ----------------
constexpr int BB  = 4;
constexpr int SS  = 2048;
constexpr int DD  = 1024;
constexpr int HH  = 16;
constexpr int DEP = 64;      // DD / HH
constexpr int FF  = 4096;
constexpr int NT  = BB * SS; // 8192 token rows

typedef __attribute__((ext_vector_type(16))) __bf16 v16bf;
typedef __attribute__((ext_vector_type(8)))  __bf16 v8bf;
typedef __attribute__((ext_vector_type(8)))  float  v8f;

__device__ __forceinline__ v8f wmma_bf16(v16bf a, v16bf b, v8f c) {
  return __builtin_amdgcn_wmma_f32_16x16x32_bf16(false, a, false, b, (short)0, c,
                                                 false, false);
}

// CDNA5 async global->LDS copy (16B per lane), tracked by ASYNCcnt.
// LDS dest = low 32 bits of generic shared pointer.
__device__ __forceinline__ void async_copy_b128(void* lds_dst, const void* gsrc) {
  unsigned           lo = (unsigned)(size_t)lds_dst;
  unsigned long long ga = (unsigned long long)gsrc;
  asm volatile("global_load_async_to_lds_b128 %0, %1, off" ::"v"(lo), "v"(ga)
               : "memory");
}
__device__ __forceinline__ void wait_async0() {
  asm volatile("s_wait_asynccnt 0x0" ::: "memory");
}

// ---------------- fp32 -> bf16 conversion (linear) ----------------
__global__ void k_cvt_bf16(const float* __restrict__ src, __bf16* __restrict__ dst,
                           int n) {
  int i = blockIdx.x * blockDim.x + threadIdx.x;
  if (i < n) dst[i] = (__bf16)src[i];
}

// fp32 [K,N] -> bf16 transposed [N,K]
__global__ void k_cvt_bf16_t(const float* __restrict__ src,
                             __bf16* __restrict__ dst, int K, int N) {
  int i = blockIdx.x * blockDim.x + threadIdx.x;
  if (i < K * N) {
    int k = i / N, n = i % N;
    dst[(size_t)n * K + k] = (__bf16)src[i];
  }
}

// ---------------- GEMM: C[M,N] = A[M,K] @ W[K,N] + bias ----------------
// W supplied TRANSPOSED (WT[N][K]) so B fragments are contiguous along K.
// 256 threads = 8 waves; block tile 64x128; wave tile 16x64 (4 WMMAs / A-frag).
// Double-buffered LDS staging via global_load_async_to_lds_b128: tile i+1 DMA
// overlaps tile i WMMAs; one barrier per K-slice.
__global__ __launch_bounds__(256) void k_gemm_bf16(
    const __bf16* __restrict__ A, const __bf16* __restrict__ WT,
    const float* __restrict__ bias, float* __restrict__ Cf,
    __bf16* __restrict__ Cb, int M, int N, int K, int relu, int transCb) {
  __shared__ __align__(16) __bf16 Alds[2][64][32];    // [buf][m][k]
  __shared__ __align__(16) __bf16 WTlds[2][128][32];  // [buf][n][k]

  const int tid  = threadIdx.x;
  const int lane = tid & 31;
  const int wave = tid >> 5;
  const int wr   = wave & 3;   // m-tile within block (4 x 16 rows)
  const int wc   = wave >> 2;  // n-half within block (2 x 64 cols)
  const int bm   = blockIdx.x * 64;
  const int bn   = blockIdx.y * 128;
  const int n16  = lane & 15;
  const int hi   = lane >> 4;

  const int ldr = tid >> 2;        // 0..63 tile row for staging
  const int ldc = (tid & 3) * 8;   // 0/8/16/24 k offset (16B chunk)

  auto issue = [&](int buf, int k0) {
    async_copy_b128(&Alds[buf][ldr][ldc], &A[(size_t)(bm + ldr) * K + k0 + ldc]);
    async_copy_b128(&WTlds[buf][ldr][ldc],
                    &WT[(size_t)(bn + ldr) * K + k0 + ldc]);
    async_copy_b128(&WTlds[buf][64 + ldr][ldc],
                    &WT[(size_t)(bn + 64 + ldr) * K + k0 + ldc]);
  };

  v8f acc[4] = {{}, {}, {}, {}};

  issue(0, 0);
  wait_async0();
  __syncthreads();

  const int niter = K / 32;
  for (int i = 0; i < niter; ++i) {
    const int b = i & 1;
    // DMA next K-slice while computing this one (ASYNCcnt overlap)
    if (i + 1 < niter) issue(b ^ 1, (i + 1) * 32);

    v16bf af;  // A fragment: lane holds row m, two contiguous 8-elem K runs
    {
      int m           = wr * 16 + n16;
      ((v8bf*)&af)[0] = *(const v8bf*)&Alds[b][m][hi * 16];
      ((v8bf*)&af)[1] = *(const v8bf*)&Alds[b][m][hi * 16 + 8];
    }
#pragma unroll
    for (int t = 0; t < 4; ++t) {
      // B fragment: lane holds col n, 16 contiguous K elems starting at hi*16
      v16bf bf = *(const v16bf*)&WTlds[b][wc * 64 + t * 16 + n16][hi * 16];
      acc[t]   = wmma_bf16(af, bf, acc[t]);
    }
    if (i + 1 < niter) {
      wait_async0();    // next slice landed
      __syncthreads();  // all waves done reading buf b / copies visible
    }
  }

#pragma unroll
  for (int t = 0; t < 4; ++t) {
#pragma unroll
    for (int v = 0; v < 8; ++v) {
      int   row = bm + wr * 16 + v + 8 * hi;  // C layout: m = v + 8*(lane/16)
      int   c   = bn + wc * 64 + t * 16 + n16;
      float xv  = acc[t][v] + bias[c];
      if (relu) xv = fmaxf(xv, 0.f);
      if (Cf) Cf[(size_t)row * N + c] = xv;
      if (Cb) {
        if (transCb) {  // bf16 output stored [N][M] (for attention V)
          Cb[(size_t)c * M + row] = (__bf16)xv;
        } else {
          Cb[(size_t)row * N + c] = (__bf16)xv;
        }
      }
    }
  }
}

// ---------------- flash attention ----------------
// One wave per (b, h, 16-query tile). Streaming softmax, causal early exit.
// Q/K fragments: contiguous vector loads. V consumed transposed VT[D][NT] so
// P.V B-fragments are contiguous 32B loads.
__global__ __launch_bounds__(32) void k_attn(
    const __bf16* __restrict__ Q, const __bf16* __restrict__ Km,
    const __bf16* __restrict__ VT, const int* __restrict__ mask,
    float* __restrict__ O) {
  __shared__ __align__(16) __bf16 Plds[16][32];

  const int lane = threadIdx.x;
  const int n16  = lane & 15;
  const int hi   = lane >> 4;

  int       blk = blockIdx.x;
  const int qt  = blk % (SS / 16); blk /= (SS / 16);
  const int h   = blk % HH;
  const int b   = blk / HH;

  const size_t rowQ = (size_t)(b * SS + qt * 16);
  const int    hc   = h * DEP;

  // Q fragments: 16 rows x depth 64 = two 16x32 A-frags (4x16B vector loads)
  v16bf aq0, aq1;
  {
    const __bf16* qrow = Q + (rowQ + n16) * DD + hc;
    ((v8bf*)&aq0)[0]   = *(const v8bf*)(qrow + hi * 16);
    ((v8bf*)&aq0)[1]   = *(const v8bf*)(qrow + 8 + hi * 16);
    ((v8bf*)&aq1)[0]   = *(const v8bf*)(qrow + 32 + hi * 16);
    ((v8bf*)&aq1)[1]   = *(const v8bf*)(qrow + 40 + hi * 16);
  }

  float mrun[8], lrun[8];
  v8f   o0 = {}, o1 = {}, o2 = {}, o3 = {};
#pragma unroll
  for (int v = 0; v < 8; ++v) { mrun[v] = -1e30f; lrun[v] = 0.f; }

  const int qg_base = qt * 16;
  const int kend    = ((qg_base + 16) + 31) & ~31;  // causal: skip keys > q_max
  const int kb      = hi * 16;

  for (int kt = 0; kt < kend; kt += 32) {
    // ---- S = Q @ K^T for 32 keys (two 16x16 col tiles); 32B K loads ----
    const __bf16* kr0 = Km + (size_t)(b * SS + kt + n16) * DD + hc;
    const __bf16* kr1 = Km + (size_t)(b * SS + kt + 16 + n16) * DD + hc;
    v16bf bk0a = *(const v16bf*)(kr0 + kb);
    v16bf bk0b = *(const v16bf*)(kr0 + 32 + kb);
    v16bf bk1a = *(const v16bf*)(kr1 + kb);
    v16bf bk1b = *(const v16bf*)(kr1 + 32 + kb);

    v8f s0 = {}, s1 = {};
    s0 = wmma_bf16(aq0, bk0a, s0);
    s0 = wmma_bf16(aq1, bk0b, s0);
    s1 = wmma_bf16(aq0, bk1a, s1);
    s1 = wmma_bf16(aq1, bk1b, s1);

    // ---- scale + causal/padding mask, streaming softmax update ----
    const int mk0 = mask[b * SS + kt + n16];
    const int mk1 = mask[b * SS + kt + 16 + n16];
#pragma unroll
    for (int v = 0; v < 8; ++v) {
      int qg  = qg_base + v + 8 * hi;
      int k0g = kt + n16;
      int k1g = kt + 16 + n16;
      s0[v] = s0[v] * 0.125f + (((k0g > qg) || (mk0 != 0)) ? -1e9f : 0.f);
      s1[v] = s1[v] * 0.125f + (((k1g > qg) || (mk1 != 0)) ? -1e9f : 0.f);

      float rm = fmaxf(s0[v], s1[v]);
      rm = fmaxf(rm, __shfl_xor(rm, 1, 32));
      rm = fmaxf(rm, __shfl_xor(rm, 2, 32));
      rm = fmaxf(rm, __shfl_xor(rm, 4, 32));
      rm = fmaxf(rm, __shfl_xor(rm, 8, 32));

      float mn    = fmaxf(mrun[v], rm);
      float alpha = __expf(mrun[v] - mn);
      mrun[v]     = mn;
      float p0    = __expf(s0[v] - mn);
      float p1    = __expf(s1[v] - mn);
      float rs    = p0 + p1;
      rs += __shfl_xor(rs, 1, 32);
      rs += __shfl_xor(rs, 2, 32);
      rs += __shfl_xor(rs, 4, 32);
      rs += __shfl_xor(rs, 8, 32);
      lrun[v] = lrun[v] * alpha + rs;
      o0[v] *= alpha; o1[v] *= alpha; o2[v] *= alpha; o3[v] *= alpha;
      s0[v] = p0;  // reuse as P
      s1[v] = p1;
    }

    // ---- transpose P (C-layout -> A-layout) via LDS ----
    __syncthreads();
#pragma unroll
    for (int v = 0; v < 8; ++v) {
      int r             = v + 8 * hi;
      Plds[r][n16]      = (__bf16)s0[v];
      Plds[r][16 + n16] = (__bf16)s1[v];
    }
    __syncthreads();
    v16bf ap;
    ((v8bf*)&ap)[0] = *(const v8bf*)&Plds[n16][hi * 16];
    ((v8bf*)&ap)[1] = *(const v8bf*)&Plds[n16][hi * 16 + 8];

    // ---- O += P @ V : V fragments contiguous 32B loads from VT[D][NT] ----
    const size_t vbase = (size_t)b * SS + kt + kb;
    v16bf bv0 = *(const v16bf*)&VT[(size_t)(hc + n16) * NT + vbase];
    v16bf bv1 = *(const v16bf*)&VT[(size_t)(hc + 16 + n16) * NT + vbase];
    v16bf bv2 = *(const v16bf*)&VT[(size_t)(hc + 32 + n16) * NT + vbase];
    v16bf bv3 = *(const v16bf*)&VT[(size_t)(hc + 48 + n16) * NT + vbase];

    o0 = wmma_bf16(ap, bv0, o0);
    o1 = wmma_bf16(ap, bv1, o1);
    o2 = wmma_bf16(ap, bv2, o2);
    o3 = wmma_bf16(ap, bv3, o3);
  }

  // ---- normalize and write attention output (fp32) ----
#pragma unroll
  for (int v = 0; v < 8; ++v) {
    float  inv = 1.0f / lrun[v];
    size_t row = rowQ + v + 8 * hi;
    float* op  = O + row * DD + hc;
    op[n16]      = o0[v] * inv;
    op[16 + n16] = o1[v] * inv;
    op[32 + n16] = o2[v] * inv;
    op[48 + n16] = o3[v] * inv;
  }
}

// ---------------- residual add + LayerNorm ----------------
__global__ __launch_bounds__(256) void k_add_ln(
    const float* __restrict__ A, const float* __restrict__ Bv,
    const float* __restrict__ gamma, const float* __restrict__ beta,
    float* __restrict__ Of, __bf16* __restrict__ Ob) {
  __shared__ float red[2][8];
  const int    row = blockIdx.x;
  const int    tid = threadIdx.x;
  const float* a   = A + (size_t)row * DD;
  const float* bb  = Bv + (size_t)row * DD;

  float vals[4];
  float s = 0.f, s2 = 0.f;
#pragma unroll
  for (int i = 0; i < 4; ++i) {
    int   c = tid + i * 256;
    float t = a[c] + bb[c];
    vals[i] = t;
    s += t;
    s2 += t * t;
  }
  for (int off = 16; off; off >>= 1) {
    s  += __shfl_down(s, off, 32);
    s2 += __shfl_down(s2, off, 32);
  }
  const int wave = tid >> 5, lane = tid & 31;
  if (lane == 0) { red[0][wave] = s; red[1][wave] = s2; }
  __syncthreads();
  if (tid == 0) {
    float ts = 0.f, ts2 = 0.f;
    for (int w = 0; w < 8; ++w) { ts += red[0][w]; ts2 += red[1][w]; }
    red[0][0] = ts;
    red[1][0] = ts2;
  }
  __syncthreads();
  const float mu   = red[0][0] / (float)DD;
  const float var  = red[1][0] / (float)DD - mu * mu;
  const float rstd = rsqrtf(var + 1e-8f);
#pragma unroll
  for (int i = 0; i < 4; ++i) {
    int   c = tid + i * 256;
    float y = gamma[c] * (vals[i] - mu) * rstd + beta[c];
    Of[(size_t)row * DD + c] = y;
    if (Ob) Ob[(size_t)row * DD + c] = (__bf16)y;
  }
}

// ---------------- host side ----------------
extern "C" void kernel_launch(void* const* d_in, const int* in_sizes, int n_in,
                              void* d_out, int out_size, void* d_ws,
                              size_t ws_size, hipStream_t stream) {
  const float* x    = (const float*)d_in[0];
  const int*   mask = (const int*)d_in[1];
  const float* wq   = (const float*)d_in[2];
  const float* bq   = (const float*)d_in[3];
  const float* wk   = (const float*)d_in[4];
  const float* bk   = (const float*)d_in[5];
  const float* wv   = (const float*)d_in[6];
  const float* bv   = (const float*)d_in[7];
  const float* w1   = (const float*)d_in[8];
  const float* b1   = (const float*)d_in[9];
  const float* w2   = (const float*)d_in[10];
  const float* b2   = (const float*)d_in[11];
  const float* g1   = (const float*)d_in[12];
  const float* be1  = (const float*)d_in[13];
  const float* g2   = (const float*)d_in[14];
  const float* be2  = (const float*)d_in[15];
  float*       out  = (float*)d_out;

  char* p     = (char*)d_ws;
  auto  carve = [&](size_t bytes) {
    char* r = p;
    p += (bytes + 255) & ~(size_t)255;
    return (void*)r;
  };

  __bf16* xbf    = (__bf16*)carve((size_t)NT * DD * 2);
  __bf16* wqT    = (__bf16*)carve((size_t)DD * DD * 2);
  __bf16* wkT    = (__bf16*)carve((size_t)DD * DD * 2);
  __bf16* wvT    = (__bf16*)carve((size_t)DD * DD * 2);
  __bf16* w1T    = (__bf16*)carve((size_t)DD * FF * 2);
  __bf16* w2T    = (__bf16*)carve((size_t)FF * DD * 2);
  __bf16* qbf    = (__bf16*)carve((size_t)NT * DD * 2);
  __bf16* kbf    = (__bf16*)carve((size_t)NT * DD * 2);
  __bf16* vtbf   = (__bf16*)carve((size_t)NT * DD * 2);  // VT[D][NT]
  float*  attnf  = (float*)carve((size_t)NT * DD * 4);
  float*  out1f  = (float*)carve((size_t)NT * DD * 4);
  __bf16* out1bf = (__bf16*)carve((size_t)NT * DD * 2);
  __bf16* hidbf  = (__bf16*)carve((size_t)NT * FF * 2);
  float*  ffnf   = (float*)carve((size_t)NT * DD * 4);

  k_cvt_bf16<<<dim3((NT * DD + 255) / 256), 256, 0, stream>>>(x, xbf, NT * DD);
  k_cvt_bf16_t<<<dim3((DD * DD + 255) / 256), 256, 0, stream>>>(wq, wqT, DD, DD);
  k_cvt_bf16_t<<<dim3((DD * DD + 255) / 256), 256, 0, stream>>>(wk, wkT, DD, DD);
  k_cvt_bf16_t<<<dim3((DD * DD + 255) / 256), 256, 0, stream>>>(wv, wvT, DD, DD);
  k_cvt_bf16_t<<<dim3((DD * FF + 255) / 256), 256, 0, stream>>>(w1, w1T, DD, FF);
  k_cvt_bf16_t<<<dim3((FF * DD + 255) / 256), 256, 0, stream>>>(w2, w2T, FF, DD);

  // QKV projections (bf16 out; V written transposed)
  dim3 gQKV(NT / 64, DD / 128);
  k_gemm_bf16<<<gQKV, 256, 0, stream>>>(xbf, wqT, bq, nullptr, qbf, NT, DD, DD, 0, 0);
  k_gemm_bf16<<<gQKV, 256, 0, stream>>>(xbf, wkT, bk, nullptr, kbf, NT, DD, DD, 0, 0);
  k_gemm_bf16<<<gQKV, 256, 0, stream>>>(xbf, wvT, bv, nullptr, vtbf, NT, DD, DD, 0, 1);

  // flash attention
  k_attn<<<dim3(BB * HH * (SS / 16)), dim3(32), 0, stream>>>(qbf, kbf, vtbf, mask,
                                                             attnf);

  // out1 = LN(x + attn)
  k_add_ln<<<dim3(NT), dim3(256), 0, stream>>>(x, attnf, g1, be1, out1f, out1bf);

  // FFN
  k_gemm_bf16<<<dim3(NT / 64, FF / 128), 256, 0, stream>>>(out1bf, w1T, b1, nullptr,
                                                           hidbf, NT, FF, DD, 1, 0);
  k_gemm_bf16<<<dim3(NT / 64, DD / 128), 256, 0, stream>>>(hidbf, w2T, b2, ffnf,
                                                           nullptr, NT, DD, FF, 0, 0);

  // out2 = LN(out1 + ffn) -> d_out
  k_add_ln<<<dim3(NT), dim3(256), 0, stream>>>(out1f, ffnf, g2, be2, out, nullptr);
}
